// KGANS_3813930959062
// MI455X (gfx1250) — matrix operand order repaired and checked by reading, not compiled, not run on hardware
//
#include <hip/hip_runtime.h>
#include <hip/hip_bf16.h>
#include <math.h>

typedef __attribute__((ext_vector_type(16))) __bf16 v16bf;
typedef __attribute__((ext_vector_type(8)))  float  v8f;

#define N_ENT 100000
#define KNB   64
#define DIM   128
#define BATCH 8192

__device__ __forceinline__ float leaky02(float x)  { return x > 0.f ? x : 0.2f * x; }
__device__ __forceinline__ float sigmoidf_(float x){ return 1.f / (1.f + expf(-x)); }

// ---------------------------------------------------------------------------
// Prep: convert Wa1[:,128:256] and Wa2 to bf16 in workspace (done each launch)
// ---------------------------------------------------------------------------
__global__ void kgans_prep(const float* __restrict__ Wa1, const float* __restrict__ Wa2,
                           __bf16* __restrict__ wa1r, __bf16* __restrict__ wa2b) {
  int i = blockIdx.x * blockDim.x + threadIdx.x;   // 0..16383
  if (i < DIM * DIM) {
    int d = i >> 7, f = i & 127;
    wa1r[i] = (__bf16)Wa1[d * 256 + 128 + f];
    wa2b[i] = (__bf16)Wa2[i];
  }
}

// ---------------------------------------------------------------------------
// 64x128 = [64xK] * Bw^T, K=N=128. bf16 WMMA, f32 accum, fused bias+ReLU,
// bf16 result to LDS.
//   A  : LDS row-major [64][128] bf16
//   Bw : global bf16 row-major [N][K]  (a[m,n] = sum_k A[m,k] * Bw[n,k])
// Schedule: wave w owns N-tile n0 = 16*w.  The 4 K-step B fragments are
// loaded ONCE into registers (32 VGPRs) and reused across all 4 M-tiles, so
// the steady state is ds_load(A) + v_wmma only.
// ---------------------------------------------------------------------------
__device__ __forceinline__ void wmma_gemm_64x128(
    const __bf16 (*A)[DIM], const __bf16* __restrict__ Bw,
    const float* __restrict__ rowbias, __bf16 (*Dst)[DIM], int tid)
{
  const int wave = tid >> 5, lane = tid & 31;
  const int lm = lane & 15, kh = lane >> 4;
  const int n0 = wave << 4;                    // 8 waves <-> 8 N-tiles
  const int n  = n0 + lm;

  // ---- hoist B fragments for all 4 K-steps ----
  v16bf bfr[4];
#pragma unroll
  for (int kt = 0; kt < 4; ++kt) {
    const int k0 = kt << 5;
#pragma unroll
    for (int v = 0; v < 8; ++v) {
      // B 32x16 striping (matches A): VGPR v holds K = (v>=4)*16 + kh*8 + (v&3)*2
      const int kb = k0 + ((v >> 2) << 4) + (kh << 3) + ((v & 3) << 1);
      bfr[kt][2 * v]     = Bw[(size_t)n * DIM + kb];
      bfr[kt][2 * v + 1] = Bw[(size_t)n * DIM + kb + 1];
    }
  }
  const float bias = rowbias ? rowbias[n] : 0.f;

  // ---- 4 M-tiles per wave, A from LDS, B resident in registers ----
#pragma unroll
  for (int mt = 0; mt < 4; ++mt) {
    const int m0 = mt << 4;
    v8f acc = {};
#pragma unroll
    for (int kt = 0; kt < 4; ++kt) {
      const int k0 = kt << 5;
      v16bf af;
#pragma unroll
      for (int v = 0; v < 8; ++v) {
        // A 16x32 striping: VGPR v holds K = (v>=4)*16 + kh*8 + (v&3)*2
        const int ka = k0 + ((v >> 2) << 4) + (kh << 3) + ((v & 3) << 1);
        af[2 * v]     = A[m0 + lm][ka];
        af[2 * v + 1] = A[m0 + lm][ka + 1];
      }
      acc = __builtin_amdgcn_wmma_f32_16x16x32_bf16(false, af, false, bfr[kt],
                                                    (short)0, acc, false, false);
    }
#pragma unroll
    for (int i = 0; i < 8; ++i) {              // C/D: VGPR i -> M = i + 8*kh
      const int m = m0 + (kh << 3) + i;
      float x = acc[i] + bias;
      Dst[m][n] = (__bf16)(x > 0.f ? x : 0.f); // fused ReLU
    }
  }
}

// ---------------------------------------------------------------------------
// One workgroup = one (branch, batch-row).  256 threads = 8 wave32.
// ---------------------------------------------------------------------------
__global__ __launch_bounds__(256) void kgans_branch(
    const int* __restrict__ u1, const int* __restrict__ u2, const int* __restrict__ cc,
    const int* __restrict__ adj_entity, const int* __restrict__ adj_relation,
    const float* __restrict__ entity_emb, const float* __restrict__ relation_emb,
    const float* __restrict__ Wa1, const float* __restrict__ Wa3,
    const float* __restrict__ Wx, const float* __restrict__ bx,
    const float* __restrict__ W1, const float* __restrict__ b1,
    const float* __restrict__ W2, const float* __restrict__ b2,
    const __bf16* __restrict__ wa1r, const __bf16* __restrict__ wa2b,
    float* __restrict__ branch_out)
{
  __shared__ float  sh_h[DIM];
  __shared__ float  sh_hA1[DIM];
  __shared__ __bf16 sh_r[KNB][DIM];   // relations; later reused as a2
  __shared__ __bf16 sh_t[KNB][DIM];   // tails
  __shared__ __bf16 sh_a1[KNB][DIM];  // GEMM1 output
  __shared__ float  sh_red[256];
  __shared__ float  sh_sc[KNB];
  __shared__ float  sh_s[DIM];
  __shared__ float  sh_p[2][2 * DIM];
  __shared__ float  sh_scale;

  const int tid = threadIdx.x;
  const int b   = blockIdx.x;
  const int br  = blockIdx.y;                 // 0=cell(c) 1=d2(u2) 2=d1(u1)
  const int* idxp = (br == 0) ? cc : ((br == 1) ? u2 : u1);
  const int e = idxp[b];

  // ---- h = lookup(entity_emb, e) with max-norm renorm ----
  float hv = 0.f;
  if (tid < DIM) {
    hv = entity_emb[(size_t)e * DIM + tid];
    sh_red[tid] = hv * hv;
  }
  __syncthreads();
  if (tid == 0) {
    float s = 0.f;
    for (int i = 0; i < DIM; ++i) s += sh_red[i];
    float n = sqrtf(s);
    sh_scale = (n > 1.f) ? 1.f / (n + 1e-7f) : 1.f;
  }
  __syncthreads();
  if (tid < DIM) sh_h[tid] = hv * sh_scale;
  __syncthreads();

  // ---- gather+renorm relations -> bf16 LDS (64 groups of 4 threads) ----
  const int g = tid >> 2, l4 = tid & 3;
  {
    const int rrow = adj_relation[(size_t)e * KNB + g];
    const float* rp = relation_emb + (size_t)rrow * DIM + l4 * 32;
    float v[32]; float ssq = 0.f;
#pragma unroll
    for (int j = 0; j < 32; ++j) { v[j] = rp[j]; ssq += v[j] * v[j]; }
    sh_red[tid] = ssq;
    __syncthreads();
    float n  = sqrtf(sh_red[g*4] + sh_red[g*4+1] + sh_red[g*4+2] + sh_red[g*4+3]);
    float sc = (n > 1.f) ? 1.f / (n + 1e-7f) : 1.f;
#pragma unroll
    for (int j = 0; j < 32; ++j) sh_r[g][l4 * 32 + j] = (__bf16)(v[j] * sc);
  }
  __syncthreads();
  // ---- gather+renorm tails -> bf16 LDS ----
  {
    const int trow = adj_entity[(size_t)e * KNB + g];
    const float* tp = entity_emb + (size_t)trow * DIM + l4 * 32;
    float v[32]; float ssq = 0.f;
#pragma unroll
    for (int j = 0; j < 32; ++j) { v[j] = tp[j]; ssq += v[j] * v[j]; }
    sh_red[tid] = ssq;
    __syncthreads();
    float n  = sqrtf(sh_red[g*4] + sh_red[g*4+1] + sh_red[g*4+2] + sh_red[g*4+3]);
    float sc = (n > 1.f) ? 1.f / (n + 1e-7f) : 1.f;
#pragma unroll
    for (int j = 0; j < 32; ++j) sh_t[g][l4 * 32 + j] = (__bf16)(v[j] * sc);
  }
  __syncthreads();

  // ---- hA1 = Wa1[:, :128] @ h  (h is broadcast over all K neighbors) ----
  if (tid < DIM) {
    const float* wrow = Wa1 + (size_t)tid * 256;
    float acc = 0.f;
#pragma unroll 8
    for (int f = 0; f < DIM; ++f) acc += wrow[f] * sh_h[f];
    sh_hA1[tid] = acc;
  }
  __syncthreads();

  // ---- GEMM1: a1 = relu(r @ Wa1r^T + hA1)   (WMMA bf16) ----
  wmma_gemm_64x128(sh_r, wa1r, sh_hA1, sh_a1, tid);
  __syncthreads();
  // ---- GEMM2: a2 = relu(a1 @ Wa2^T)         (WMMA bf16, into sh_r) ----
  wmma_gemm_64x128(sh_a1, wa2b, nullptr, sh_r, tid);
  __syncthreads();

  // ---- scores: sigmoid(Wa3 . a2) per neighbor ----
  {
    const float* w3 = Wa3 + l4 * 32;
    const __bf16* ar = &sh_r[g][l4 * 32];
    float acc = 0.f;
#pragma unroll
    for (int j = 0; j < 32; ++j) acc += (float)ar[j] * w3[j];
    sh_red[tid] = acc;
  }
  __syncthreads();
  if (tid < KNB) {
    float s = sh_red[tid*4] + sh_red[tid*4+1] + sh_red[tid*4+2] + sh_red[tid*4+3];
    sh_sc[tid] = sigmoidf_(s);
  }
  __syncthreads();
  // ---- softmax over K=64 ----
  if (tid < KNB) {
    float mx = -1e30f;
    for (int j = 0; j < KNB; ++j) mx = fmaxf(mx, sh_sc[j]);
    sh_red[tid] = expf(sh_sc[tid] - mx);
  }
  __syncthreads();
  if (tid < KNB) {
    float s = 0.f;
    for (int j = 0; j < KNB; ++j) s += sh_red[j];
    sh_sc[tid] = sh_red[tid] / s;
  }
  __syncthreads();

  // ---- s = sum_k w[k] * t[k,:] ----
  if (tid < DIM) {
    float acc = 0.f;
#pragma unroll 8
    for (int k = 0; k < KNB; ++k) acc += sh_sc[k] * (float)sh_t[k][tid];
    sh_s[tid] = acc;
  }
  __syncthreads();

  // ---- per-head projection, leaky, hrep combine ----
  {
    const int ee = tid & (DIM - 1);           // tid = h*128 + e
    const float* wx = Wx + (size_t)tid * DIM; // Wx[h][e][:]
    float acc = bx[tid];
#pragma unroll 8
    for (int d = 0; d < DIM; ++d) acc += sh_s[d] * wx[d];
    const float vec = leaky02(acc);
    const float hr  = sh_h[ee];
    sh_p[0][tid] = hr + vec;
    sh_p[1][tid] = hr * vec;
  }
  __syncthreads();

  // ---- agg = leaky(p0 @ W1^T + b1) + leaky(p1 @ W2^T + b2); emit [agg | h] ----
  {
    const float* w1 = W1 + (size_t)tid * 256;
    const float* w2 = W2 + (size_t)tid * 256;
    float a1 = b1[tid], a2 = b2[tid];
#pragma unroll 4
    for (int j = 0; j < 256; ++j) { a1 += w1[j] * sh_p[0][j]; a2 += w2[j] * sh_p[1][j]; }
    float* outrow = branch_out + ((size_t)br * BATCH + b) * 384;
    outrow[tid] = leaky02(a1) + leaky02(a2);
    if (tid < DIM) outrow[256 + tid] = sh_h[tid];
  }
}

// ---------------------------------------------------------------------------
// Final: out[b] = sigmoid( sum_i max(d1,d2)*cell ).  One wave32 per row.
// ---------------------------------------------------------------------------
__global__ __launch_bounds__(256) void kgans_combine(const float* __restrict__ bo,
                                                     float* __restrict__ out) {
  const int wave = threadIdx.x >> 5, lane = threadIdx.x & 31;
  const int row  = blockIdx.x * 8 + wave;
  const float* cell = bo + (size_t)row * 384;                       // br 0 (c)
  const float* d2p  = bo + ((size_t)BATCH + row) * 384;             // br 1 (u2)
  const float* d1p  = bo + ((size_t)2 * BATCH + row) * 384;         // br 2 (u1)
  float acc = 0.f;
  for (int i = lane; i < 384; i += 32)
    acc += fmaxf(d1p[i], d2p[i]) * cell[i];
#pragma unroll
  for (int off = 16; off; off >>= 1) acc += __shfl_xor(acc, off, 32);
  if (lane == 0) out[row] = sigmoidf_(acc);
}

extern "C" void kernel_launch(void* const* d_in, const int* in_sizes, int n_in,
                              void* d_out, int out_size, void* d_ws, size_t ws_size,
                              hipStream_t stream) {
  (void)in_sizes; (void)n_in; (void)out_size; (void)ws_size;
  const int*   u1    = (const int*)d_in[0];
  const int*   u2    = (const int*)d_in[1];
  const int*   c     = (const int*)d_in[2];
  const int*   adj_e = (const int*)d_in[3];
  const int*   adj_r = (const int*)d_in[4];
  const float* ent   = (const float*)d_in[5];
  const float* rel   = (const float*)d_in[6];
  const float* Wa1   = (const float*)d_in[7];
  const float* Wa2   = (const float*)d_in[8];
  const float* Wa3   = (const float*)d_in[9];
  const float* Wx    = (const float*)d_in[10];
  const float* bx    = (const float*)d_in[11];
  const float* W1    = (const float*)d_in[12];
  const float* b1    = (const float*)d_in[13];
  const float* W2    = (const float*)d_in[14];
  const float* b2    = (const float*)d_in[15];
  float* out = (float*)d_out;

  char* ws = (char*)d_ws;
  float* branch_out = (float*)ws;                               // [3][B][384] f32
  size_t off = (size_t)3 * BATCH * 384 * sizeof(float);
  __bf16* wa1r = (__bf16*)(ws + off); off += (size_t)DIM * DIM * sizeof(__bf16);
  __bf16* wa2b = (__bf16*)(ws + off);

  hipLaunchKernelGGL(kgans_prep, dim3(64), dim3(256), 0, stream, Wa1, Wa2, wa1r, wa2b);
  hipLaunchKernelGGL(kgans_branch, dim3(BATCH, 3), dim3(256), 0, stream,
                     u1, u2, c, adj_e, adj_r, ent, rel,
                     Wa1, Wa3, Wx, bx, W1, b1, W2, b2, wa1r, wa2b, branch_out);
  hipLaunchKernelGGL(kgans_combine, dim3(BATCH / 8), dim3(256), 0, stream,
                     branch_out, out);
}
